// AttentionBlockConv_55121610276882
// MI455X (gfx1250) — matrix operand order repaired and checked by reading, not compiled, not run on hardware
//
#include <hip/hip_runtime.h>
#include <math.h>

// ---------------------------------------------------------------------------
// AttentionBlockConv for MI455X (gfx1250, wave32).
// Fused flash-style global attention (2 x 16384 x 4096, C=8):
//   - scores via v_wmma_f32_16x16x32_bf16 (K=8 zero-padded to 32)
//   - P = exp(s-max) staged through LDS, second WMMA accumulates P*phi with a
//     ones-column at N=8 so the softmax denominator falls out of the same WMMA
//   - all WMMA operands pre-packed to bf16 once, so inner loops are pure
//     global_load_b128 -> v_wmma (no per-iteration cvt VALU, no exec branches)
// Never materializes the 512MB score matrix (~1.5GB HBM traffic avoided).
// ---------------------------------------------------------------------------

#define DEV __device__ __forceinline__

typedef __attribute__((ext_vector_type(16))) __bf16 v16bf;
typedef __attribute__((ext_vector_type(8)))  float  v8f;

union VReg {
  v16bf v;
  unsigned short s[16];
  uint4 q[2];
};

DEV unsigned short f2bf(float f) {            // f32 -> bf16 (RNE)
  unsigned u = __float_as_uint(f);
  unsigned r = u + 0x7fffu + ((u >> 16) & 1u);
  return (unsigned short)(r >> 16);
}
DEV unsigned fkey(float x) {                  // order-preserving f32 -> u32
  unsigned u = __float_as_uint(x);
  return (u & 0x80000000u) ? ~u : (u | 0x80000000u);
}
DEV float fdec(unsigned k) {
  unsigned u = (k & 0x80000000u) ? (k ^ 0x80000000u) : ~k;
  return __uint_as_float(u);
}

// ------------------------------- elementwise -------------------------------

__global__ void bn_k(const float* __restrict__ x, const float* __restrict__ g,
                     const float* __restrict__ be, const float* __restrict__ mu,
                     const float* __restrict__ va, float* __restrict__ out,
                     long n, int C) {
  long idx = (long)blockIdx.x * blockDim.x + threadIdx.x;
  if (idx >= n) return;
  int c = (int)(idx % C);
  out[idx] = g[c] * (x[idx] - mu[c]) * rsqrtf(va[c] + 1e-3f) + be[c];
}

__global__ void maxpool2_k(const float* __restrict__ in, float* __restrict__ out,
                           int B, int H, int W, int C) {
  long idx = (long)blockIdx.x * blockDim.x + threadIdx.x;
  int OH = H >> 1, OW = W >> 1;
  long total = (long)B * OH * OW * C;
  if (idx >= total) return;
  int c = (int)(idx % C); long t = idx / C;
  int w = (int)(t % OW); t /= OW;
  int h = (int)(t % OH); t /= OH;
  int b = (int)t;
  const float* p0 = in + (((long)(b * H + 2 * h) * W + 2 * w) * C) + c;
  const float* p1 = p0 + (long)W * C;
  out[idx] = fmaxf(fmaxf(p0[0], p0[C]), fmaxf(p1[0], p1[C]));
}

// ------------------------------ direct conv --------------------------------

__global__ void conv2d_k(const float* __restrict__ in, const float* __restrict__ wgt,
                         const float* __restrict__ bias, float* __restrict__ out,
                         int Bn, int H, int W, int Cin, int Cout, int KH, int KW,
                         int padh, int padw, int act, float postscale,
                         int out_choff, int out_cstride) {
  long idx = (long)blockIdx.x * blockDim.x + threadIdx.x;
  long total = (long)Bn * H * W * Cout;
  if (idx >= total) return;
  int co = (int)(idx % Cout); long t = idx / Cout;
  int x = (int)(t % W); t /= W;
  int y = (int)(t % H); t /= H;
  int b = (int)t;
  float acc = bias[co];
  for (int kh = 0; kh < KH; ++kh) {
    int ih = y + kh - padh;
    if (ih < 0 || ih >= H) continue;
    for (int kw = 0; kw < KW; ++kw) {
      int iw = x + kw - padw;
      if (iw < 0 || iw >= W) continue;
      const float* ip = in + ((long)(b * H + ih) * W + iw) * Cin;
      const float* wp = wgt + ((long)(kh * KW + kw) * Cin) * Cout + co;
      for (int ci = 0; ci < Cin; ++ci)
        acc = fmaf(ip[ci], wp[(long)ci * Cout], acc);
    }
  }
  if (act) acc = fmaxf(acc, 0.f);
  acc *= postscale;
  out[((long)(b * H + y) * W + x) * out_cstride + out_choff + co] = acc;
}

// ------------------------- bf16 operand pre-packing ------------------------
// fbf/gbf: row-major bf16 copies (one b128 load = one 16-lane operand row).
// bphi:    fully assembled 32x16 B-operand per m-pair tile: cols 0..7 = phi
//          channels, col 8 = ones (z column), cols 9..15 = 0.  Every lane
//          loads its own 32B slice unconditionally.

__global__ void pack_attn_k(const float* __restrict__ f, const float* __restrict__ g,
                            const float* __restrict__ phi,
                            unsigned short* __restrict__ fbf,
                            unsigned short* __restrict__ gbf,
                            unsigned short* __restrict__ bphi,
                            unsigned short* __restrict__ zero8) {
  long idx = (long)blockIdx.x * blockDim.x + threadIdx.x;
  const long nf = 2L * 16384 * 8;
  if (idx < nf) fbf[idx] = f2bf(f[idx]);
  const long ng = 2L * 4096 * 8;
  if (idx < ng) gbf[idx] = f2bf(g[idx]);
  const long nb = 2L * 128 * 32 * 16;           // (b, mp, lane, i)
  if (idx < nb) {
    int i = (int)(idx & 15); long t = idx >> 4;
    int lane = (int)(t & 31); t >>= 5;
    int mp = (int)(t & 127); t >>= 7;
    int b = (int)t;
    int half = lane >> 4, N = lane & 15;
    int m = mp * 32 + half * 16 + i;            // K = half*16 + i
    float val = (N < 8) ? phi[((size_t)b * 4096 + m) * 8 + N]
                        : (N == 8 ? 1.0f : 0.0f);
    bphi[idx] = f2bf(val);
  }
  if (idx < 16) zero8[idx] = 0;
}

// ------------------------------- attention ---------------------------------

__global__ void attn_max_k(const unsigned short* __restrict__ fbf,
                           const unsigned short* __restrict__ gbf,
                           const unsigned short* __restrict__ zero8,
                           unsigned* __restrict__ mk) {
  const int lane = threadIdx.x & 31;
  const int wv   = threadIdx.x >> 5;
  const int half = lane >> 4;
  const int l16  = lane & 15;
  const int b    = blockIdx.y;
  const int nbase = (blockIdx.x * 4 + wv) * 16;

  VReg a;
  a.q[0] = *(const uint4*)(half ? zero8
                                : fbf + ((size_t)b * 16384 + nbase + l16) * 8);
  a.q[1] = make_uint4(0u, 0u, 0u, 0u);

  const unsigned short* gp = half ? zero8 : gbf + ((size_t)b * 4096 + l16) * 8;
  const size_t gstep = half ? 0 : 16 * 8;       // one 16-row m-tile per step

  VReg bb;
  bb.q[1] = make_uint4(0u, 0u, 0u, 0u);
  float vmax = -3.4e38f;
#pragma unroll 4
  for (int mt = 0; mt < 256; ++mt) {
    bb.q[0] = *(const uint4*)(gp + (size_t)mt * gstep);
    v8f c = {0.f, 0.f, 0.f, 0.f, 0.f, 0.f, 0.f, 0.f};
    c = __builtin_amdgcn_wmma_f32_16x16x32_bf16(false, a.v, false, bb.v,
                                                (short)0, c, false, false);
#pragma unroll
    for (int i = 0; i < 8; ++i) vmax = fmaxf(vmax, c[i]);
  }
#pragma unroll
  for (int off = 16; off > 0; off >>= 1)
    vmax = fmaxf(vmax, __shfl_xor(vmax, off, 32));
  if (lane == 0) atomicMax(&mk[b], fkey(vmax));
}

__global__ void attn_av_k(const unsigned short* __restrict__ fbf,
                          const unsigned short* __restrict__ gbf,
                          const unsigned short* __restrict__ bphi,
                          const unsigned short* __restrict__ zero8,
                          const unsigned* __restrict__ mk,
                          float* __restrict__ apun, float* __restrict__ zpart) {
  __shared__ __align__(16) unsigned short sP[4][16][32];  // per-wave P tiles
  const int lane = threadIdx.x & 31;
  const int wv   = threadIdx.x >> 5;
  const int half = lane >> 4;
  const int l16  = lane & 15;
  const int b    = blockIdx.y;
  const int ntile = blockIdx.x * 4 + wv;
  const int nbase = ntile * 16;
  const float Mb = fdec(mk[b]);

  VReg a;
  a.q[0] = *(const uint4*)(half ? zero8
                                : fbf + ((size_t)b * 16384 + nbase + l16) * 8);
  a.q[1] = make_uint4(0u, 0u, 0u, 0u);

  const unsigned short* gp = half ? zero8 : gbf + ((size_t)b * 4096 + l16) * 8;
  const size_t gstep = half ? 0 : 16 * 8;
  const unsigned short* bpp = bphi + (((size_t)b * 128 * 32) + lane) * 16;

  v8f apacc = {0.f, 0.f, 0.f, 0.f, 0.f, 0.f, 0.f, 0.f};
  VReg bb;
  bb.q[1] = make_uint4(0u, 0u, 0u, 0u);

  for (int mp = 0; mp < 128; ++mp) {
    // two 16x16 score tiles -> exp -> LDS (bf16, A-layout source)
#pragma unroll
    for (int t = 0; t < 2; ++t) {
      bb.q[0] = *(const uint4*)(gp + (size_t)(mp * 2 + t) * gstep);
      v8f c = {0.f, 0.f, 0.f, 0.f, 0.f, 0.f, 0.f, 0.f};
      c = __builtin_amdgcn_wmma_f32_16x16x32_bf16(false, a.v, false, bb.v,
                                                  (short)0, c, false, false);
#pragma unroll
      for (int i = 0; i < 8; ++i) {
        float p = __expf(c[i] - Mb);
        sP[wv][i + 8 * half][t * 16 + l16] = f2bf(p);   // [M][K]
      }
    }
    asm volatile("s_wait_dscnt 0" ::: "memory");  // LDS RAW within wave

    // A2 = P tile (16 x 32 bf16) in A layout
    VReg a2;
    const int k1 = half * 16, k2 = half * 16 + 8;
    a2.q[0] = *(const uint4*)&sP[wv][l16][k1];
    a2.q[1] = *(const uint4*)&sP[wv][l16][k2];

    // B = phi (pre-packed, ones column at N=8)
    VReg bp;
    const unsigned short* bsrc = bpp + (size_t)mp * 32 * 16;
    bp.q[0] = *(const uint4*)bsrc;
    bp.q[1] = *(const uint4*)(bsrc + 8);
    apacc = __builtin_amdgcn_wmma_f32_16x16x32_bf16(false, a2.v, false, bp.v,
                                                    (short)0, apacc, false, false);
  }

  float zsum = 0.f;
#pragma unroll
  for (int i = 0; i < 8; ++i) {
    int n = nbase + i + 8 * half;
    if (l16 < 8)       apun[((size_t)b * 16384 + n) * 8 + l16] = apacc[i];
    else if (l16 == 8) zsum += apacc[i];
  }
  zsum += __shfl_xor(zsum, 16, 32);               // combine the two halves
  if (lane == 8) zpart[b * 1024 + ntile] = zsum;  // deterministic partials
}

__global__ void reduce_z_k(const float* __restrict__ zpart, float* __restrict__ Z) {
  __shared__ float sm[256];
  int b = blockIdx.x;
  float s = 0.f;
  for (int i = threadIdx.x; i < 1024; i += 256) s += zpart[b * 1024 + i];
  sm[threadIdx.x] = s; __syncthreads();
  for (int o = 128; o > 0; o >>= 1) {
    if ((int)threadIdx.x < o) sm[threadIdx.x] += sm[threadIdx.x + o];
    __syncthreads();
  }
  if (threadIdx.x == 0) Z[b] = sm[0];
}

// finalize: ap = apun/Z ; ap2 = last(ap) ; y = inp + sigma*ap2
__global__ void finalize_k(const float* __restrict__ inp, const float* __restrict__ apun,
                           const float* __restrict__ Z, const float* __restrict__ lw,
                           const float* __restrict__ lb, const float* __restrict__ sigma,
                           float* __restrict__ y) {
  long idx = (long)blockIdx.x * blockDim.x + threadIdx.x;  // b*16384 + hw
  if (idx >= 2L * 16384) return;
  int b = (int)(idx >> 14);
  float zi = 1.f / Z[b];
  float ap[8];
#pragma unroll
  for (int c = 0; c < 8; ++c) ap[c] = apun[idx * 8 + c] * zi;
  float sg = sigma[0];
#pragma unroll
  for (int co = 0; co < 8; ++co) {
    float acc = lb[co];
#pragma unroll
    for (int ci = 0; ci < 8; ++ci) acc = fmaf(lw[ci * 8 + co], ap[ci], acc);
    y[idx * 8 + co] = inp[idx * 8 + co] + sg * acc;
  }
}

__global__ void init_stats_k(unsigned* mk) {
  if (threadIdx.x < 2) mk[threadIdx.x] = 0u;
}

// --------------------------------- launch ----------------------------------

extern "C" void kernel_launch(void* const* d_in, const int* in_sizes, int n_in,
                              void* d_out, int out_size, void* d_ws, size_t ws_size,
                              hipStream_t stream) {
  (void)in_sizes; (void)n_in; (void)out_size; (void)ws_size;
  const float* const* I = (const float* const*)d_in;
  // setup_inputs() flattening: x, then params in dict order (bn tuples as g,b,m,v)
  const float* x      = I[0];
  const float* bn1[4] = {I[1], I[2], I[3], I[4]};
  const float *cb1w = I[5],  *cb1b = I[6];
  const float* bn2[4] = {I[7], I[8], I[9], I[10]};
  const float *cb2w = I[11], *cb2b = I[12];
  const float* bn3[4] = {I[13], I[14], I[15], I[16]};
  const float *cb3w = I[17], *cb3b = I[18];
  const float* bn4[4] = {I[19], I[20], I[21], I[22]};
  const float *cb4w = I[23], *cb4b = I[24];
  const float *ampw = I[25], *ampb = I[26];
  const float *phw  = I[27], *phb  = I[28];
  const float *qw = I[29], *qb = I[30];
  const float *kw = I[31], *kb = I[32];
  const float *vw = I[33], *vb = I[34];
  const float *lw = I[35], *lb = I[36];
  const float *ow = I[37], *ob = I[38];
  const float *sigma = I[39];

  const int B = 2, H = 128, W = 128;
  const long HW = (long)H * W;

  // workspace layout
  char* ws = (char*)d_ws;
  const size_t SZ_INP  = (size_t)B * HW * 8 * 4;
  const size_t SZ_F    = SZ_INP;
  const size_t SZ_G    = (size_t)B * 4096 * 8 * 4;
  const size_t SZ_APUN = SZ_INP;
  const size_t SZ_Y    = SZ_INP;
  const size_t SZ_ZP   = (size_t)B * 1024 * 4;
  const size_t SZ_FBF  = (size_t)B * HW * 8 * 2;
  const size_t SZ_GBF  = (size_t)B * 4096 * 8 * 2;
  const size_t SZ_BPHI = (size_t)B * 128 * 32 * 16 * 2;
  const size_t SZ_Z8   = 256;
  const size_t SZ_PING = (size_t)B * HW * 64 * 4;

  unsigned* mk  = (unsigned*)ws;            // 2 x u32 ordered max
  float* Zp     = (float*)(ws + 16);        // 2 x f32
  size_t off = 256;
  float* inp  = (float*)(ws + off); off += SZ_INP;
  float* fq   = (float*)(ws + off); off += SZ_F;
  float* gk   = (float*)(ws + off); off += SZ_G;
  float* phi  = (float*)(ws + off); off += SZ_G;
  float* apun = (float*)(ws + off); off += SZ_APUN;
  float* yb   = (float*)(ws + off); off += SZ_Y;
  float* zpt  = (float*)(ws + off); off += SZ_ZP;
  unsigned short* fbf  = (unsigned short*)(ws + off); off += SZ_FBF;
  unsigned short* gbf  = (unsigned short*)(ws + off); off += SZ_GBF;
  unsigned short* bphi = (unsigned short*)(ws + off); off += SZ_BPHI;
  unsigned short* z8   = (unsigned short*)(ws + off); off += SZ_Z8;
  float* ping = (float*)(ws + off); off += SZ_PING;
  float* pong = (float*)(ws + off);

  auto blocks = [](long n) { return (unsigned)((n + 255) / 256); };
  const float PI = 3.14159265358979323846f;

  // ---- ConvBlock ----
  long n;
  n = (long)B*HW*4;  bn_k<<<blocks(n),256,0,stream>>>(x, bn1[0],bn1[1],bn1[2],bn1[3], ping, n, 4);
  n = (long)B*HW*20; conv2d_k<<<blocks(n),256,0,stream>>>(ping, cb1w, cb1b, pong, B,H,W, 4,20, 8,8, 3,3, 1, 1.f, 0,20);
  bn_k<<<blocks(n),256,0,stream>>>(pong, bn2[0],bn2[1],bn2[2],bn2[3], ping, n, 20);
  n = (long)B*HW*30; conv2d_k<<<blocks(n),256,0,stream>>>(ping, cb2w, cb2b, pong, B,H,W, 20,30, 5,5, 2,2, 1, 1.f, 0,30);
  bn_k<<<blocks(n),256,0,stream>>>(pong, bn3[0],bn3[1],bn3[2],bn3[3], ping, n, 30);
  n = (long)B*HW*40; conv2d_k<<<blocks(n),256,0,stream>>>(ping, cb3w, cb3b, pong, B,H,W, 30,40, 5,5, 2,2, 1, 1.f, 0,40);
  bn_k<<<blocks(n),256,0,stream>>>(pong, bn4[0],bn4[1],bn4[2],bn4[3], ping, n, 40);
  n = (long)B*HW*50; conv2d_k<<<blocks(n),256,0,stream>>>(ping, cb4w, cb4b, pong, B,H,W, 40,50, 5,5, 2,2, 1, 1.f, 0,50);
  // amp||pi*phase -> inp (8ch)
  n = (long)B*HW*4;
  conv2d_k<<<blocks(n),256,0,stream>>>(pong, ampw, ampb, inp, B,H,W, 50,4, 3,3, 1,1, 1, 1.f, 0,8);
  conv2d_k<<<blocks(n),256,0,stream>>>(pong, phw,  phb,  inp, B,H,W, 50,4, 3,3, 1,1, 0, PI,  4,8);

  // ---- q/k/v (1x1) + maxpool on k,v ----
  n = (long)B*HW*8;
  conv2d_k<<<blocks(n),256,0,stream>>>(inp, qw, qb, fq,   B,H,W, 8,8, 1,1, 0,0, 0, 1.f, 0,8);
  conv2d_k<<<blocks(n),256,0,stream>>>(inp, kw, kb, ping, B,H,W, 8,8, 1,1, 0,0, 0, 1.f, 0,8);
  maxpool2_k<<<blocks((long)B*4096*8),256,0,stream>>>(ping, gk, B,H,W,8);
  conv2d_k<<<blocks(n),256,0,stream>>>(inp, vw, vb, ping, B,H,W, 8,8, 1,1, 0,0, 0, 1.f, 0,8);
  maxpool2_k<<<blocks((long)B*4096*8),256,0,stream>>>(ping, phi, B,H,W,8);

  // ---- fused attention (WMMA) ----
  init_stats_k<<<1,32,0,stream>>>(mk);
  pack_attn_k<<<blocks(2L*16384*8),256,0,stream>>>(fq, gk, phi, fbf, gbf, bphi, z8);
  dim3 agrid(256, B);
  attn_max_k<<<agrid,128,0,stream>>>(fbf, gbf, z8, mk);
  attn_av_k<<<agrid,128,0,stream>>>(fbf, gbf, bphi, z8, mk, apun, zpt);
  reduce_z_k<<<B,256,0,stream>>>(zpt, Zp);
  finalize_k<<<blocks((long)B*HW),256,0,stream>>>(inp, apun, Zp, lw, lb, sigma, yb);

  // ---- out conv 6x6 -> d_out ----
  n = (long)B*HW*4;
  conv2d_k<<<blocks(n),256,0,stream>>>(yb, ow, ob, (float*)d_out, B,H,W, 8,4, 6,6, 2,2, 0, 1.f, 0,4);
}
